// GCNModel_21328807592519
// MI455X (gfx1250) — matrix operand order repaired and checked by reading, not compiled, not run on hardware
//
#include <hip/hip_runtime.h>
#include <hip/hip_bf16.h>

typedef __attribute__((ext_vector_type(16))) _Float16 v16h;
typedef __attribute__((ext_vector_type(4)))  _Float16 v4h;
typedef __attribute__((ext_vector_type(8)))  float    v8f;

#define HD 128           // feature / hidden width (F == H == 128)
#define NEG_SLOPE 0.01f

// ---------------- degree / normalization ----------------
__global__ void k_deg_init(float* deg, int n) {
    int i = blockIdx.x * blockDim.x + threadIdx.x;
    if (i < n) deg[i] = 1.0f;                      // self-loop contributes 1
}

__global__ void k_deg_count(const int* __restrict__ col, float* deg, int e) {
    int i = blockIdx.x * blockDim.x + threadIdx.x;
    if (i < e) atomicAdd(&deg[col[i]], 1.0f);
}

__global__ void k_rsqrt_inplace(float* deg, int n) {
    int i = blockIdx.x * blockDim.x + threadIdx.x;
    if (i < n) deg[i] = rsqrtf(deg[i]);            // deg >= 1 always (self loop)
}

// ---------------- precision convert (x -> f16, vectorized x4) ----------------
__global__ void k_f32_to_f16_v4(const float* __restrict__ src, _Float16* __restrict__ dst,
                                long long total4) {
    long long i = (long long)blockIdx.x * blockDim.x + threadIdx.x;
    if (i >= total4) return;
    const float4 v = ((const float4*)src)[i];
    v4h p = {(_Float16)v.x, (_Float16)v.y, (_Float16)v.z, (_Float16)v.w};
    ((v4h*)dst)[i] = p;
}

// ---------------- pack W (f32) into WMMA B-fragment order (f16) ----------------
// Layout: wpk[(((ct*4 + ks)*32 + lane)*16 + elem]
//   elem e of lane's fragment = B[k, col], k = ks*32 + 16*(lane>>4) + e,
//   col = ct*16 + (lane&15)   (contiguous k order, same as the A fragment)
__global__ void k_pack_w(const float* __restrict__ W, _Float16* __restrict__ wpk) {
    int idx = blockIdx.x * blockDim.x + threadIdx.x;      // 16384 total
    if (idx >= HD * HD) return;
    int elem = idx & 15;
    int lane = (idx >> 4) & 31;
    int ks   = (idx >> 9) & 3;
    int ct   = idx >> 11;                                  // 0..7
    int kb   = ks * 32 + 16 * (lane >> 4) + elem;
    int col  = ct * 16 + (lane & 15);
    wpk[idx] = (_Float16)W[kb * HD + col];
}

// ---------------- WMMA GEMM: C[n x 128] = A[n x 128](f16) * W(packed f16), f32 acc ----
// blockDim = 128 (4 waves). Block owns 16 rows; wave wv owns col tiles {2wv, 2wv+1}.
// A fragment = two contiguous 16B runs per lane -> b128 loads.
// B fragment = one contiguous 32B run per lane from the packed buffer.
// C store: fast path (full tile) = 1 address + 8 immediate-offset b32 stores.
__global__ void k_gemm_wmma(const _Float16* __restrict__ A, const _Float16* __restrict__ Bpk,
                            float* __restrict__ C, int nrows) {
    const int tid  = threadIdx.x;
    const int wave = tid >> 5;
    const int lane = tid & 31;
    const int m    = lane & 15;
    const int half = lane >> 4;
    const int rowBase = blockIdx.x * 16;
    int row = rowBase + m;
    if (row >= nrows) row = nrows - 1;      // clamp (harmless redundant read)

    v16h afrag[4];
#pragma unroll
    for (int ks = 0; ks < 4; ++ks) {
        const _Float16* a0 = A + (size_t)row * HD + ks * 32 + 16 * half;  // K=k0+16h..+7
        const _Float16* a1 = a0 + 8;                                      // K=k0+16h+8..+15
#pragma unroll
        for (int j = 0; j < 8; ++j) {
            afrag[ks][j]     = a0[j];
            afrag[ks][8 + j] = a1[j];
        }
    }

    const bool fullTile = (rowBase + 16) <= nrows;

#pragma unroll
    for (int ct = 0; ct < 2; ++ct) {
        const int wct = wave * 2 + ct;                 // global col tile 0..7
        const int colBase = wct * 16;
        v8f acc = {};
#pragma unroll
        for (int ks = 0; ks < 4; ++ks) {
            const v16h b = *(const v16h*)(Bpk + ((((size_t)wct * 4) + ks) * 32 + lane) * 16);
            acc = __builtin_amdgcn_wmma_f32_16x16x32_f16(
                /*neg_a=*/false, afrag[ks], /*neg_b=*/false, b,
                /*c_mod=*/(short)0, acc, /*reuse_a=*/false, /*reuse_b=*/false);
        }
        // f32 C/D layout: VGPR v -> M = v + 8*half, N = lane%16
        if (fullTile) {
            float* cp = C + (size_t)(rowBase + 8 * half) * HD + colBase + m;
#pragma unroll
            for (int v = 0; v < 8; ++v)
                cp[(size_t)v * HD] = acc[v];           // immediate-offset stores
        } else {
#pragma unroll
            for (int v = 0; v < 8; ++v) {
                const int outRow = rowBase + v + 8 * half;
                if (outRow < nrows)
                    C[(size_t)outRow * HD + colBase + m] = acc[v];
            }
        }
    }
}

// ---------------- aggregation ----------------
// acc = hw * dinv^2 (self loop) + bias   (handles the (A+I) self edge without atomics)
__global__ void k_init_acc(const float* __restrict__ hw, const float* __restrict__ dinv,
                           const float* __restrict__ bias, float* __restrict__ acc,
                           long long total4) {
    long long idx = (long long)blockIdx.x * blockDim.x + threadIdx.x;
    if (idx >= total4) return;
    int node = (int)(idx >> 5);                  // 32 float4 per 128-wide row
    int c4   = (int)(idx & 31);
    float d  = dinv[node];
    float dd = d * d;
    const float4 v = ((const float4*)hw)[idx];
    const float4 b = ((const float4*)bias)[c4];
    float4 r = {v.x * dd + b.x, v.y * dd + b.y, v.z * dd + b.z, v.w * dd + b.w};
    ((float4*)acc)[idx] = r;
}

// one wave per edge, each lane owns 4 channels: float4 gather from hw[row],
// 4 f32 atomics into acc[col]. Edge index forced wave-uniform (scalar loads).
__global__ void k_scatter_edges(const float* __restrict__ hw, const int* __restrict__ rows,
                                const int* __restrict__ cols, const float* __restrict__ dinv,
                                float* __restrict__ acc, int e) {
    int wid = __builtin_amdgcn_readfirstlane(
        (int)(blockIdx.x * 8u + (threadIdx.x >> 5)));   // blockDim == 256 -> 8 waves
    if (wid >= e) return;
    int lane = threadIdx.x & 31;
    int r = rows[wid];
    int c = cols[wid];
    float nr = dinv[r] * dinv[c];
    const int c0 = lane << 2;
    const float4 v = *(const float4*)(hw + (size_t)r * HD + c0);
    float* dst = acc + (size_t)c * HD + c0;
    atomicAdd(dst + 0, v.x * nr);
    atomicAdd(dst + 1, v.y * nr);
    atomicAdd(dst + 2, v.z * nr);
    atomicAdd(dst + 3, v.w * nr);
}

// leaky-ReLU fused with next layer's f16 conversion (saves one full pass over h)
__global__ void k_leaky_dual(const float* __restrict__ acc, float* __restrict__ h,
                             _Float16* __restrict__ h16, long long total4) {
    long long i = (long long)blockIdx.x * blockDim.x + threadIdx.x;
    if (i >= total4) return;
    const float4 v = ((const float4*)acc)[i];
    float4 r;
    r.x = v.x > 0.0f ? v.x : NEG_SLOPE * v.x;
    r.y = v.y > 0.0f ? v.y : NEG_SLOPE * v.y;
    r.z = v.z > 0.0f ? v.z : NEG_SLOPE * v.z;
    r.w = v.w > 0.0f ? v.w : NEG_SLOPE * v.w;
    ((float4*)h)[i] = r;
    v4h p = {(_Float16)r.x, (_Float16)r.y, (_Float16)r.z, (_Float16)r.w};
    ((v4h*)h16)[i] = p;
}

// ---------------- readout ----------------
__global__ void k_zero128(float* g) {
    int i = threadIdx.x;
    if (i < HD) g[i] = 0.0f;
}

#define CHUNK 256
__global__ void k_colsum(const float* __restrict__ h, float* __restrict__ gsum, int nrows) {
    int c  = threadIdx.x;                 // 128 threads = 1 channel each
    int r0 = blockIdx.x * CHUNK;
    int r1 = r0 + CHUNK; if (r1 > nrows) r1 = nrows;
    float s = 0.0f;
    for (int r = r0; r < r1; ++r) s += h[(size_t)r * HD + c];
    atomicAdd(&gsum[c], s);
}

__global__ void k_finalize(const float* __restrict__ gsum, const float* __restrict__ lin_w,
                           const float* __restrict__ lin_b, const float* __restrict__ u,
                           const float* __restrict__ w, float* __restrict__ out, float invN) {
    __shared__ float red[4];
    int tid = threadIdx.x;                // 128 threads, 4 waves
    float s = gsum[tid] * invN * lin_w[tid];
    for (int off = 16; off > 0; off >>= 1) s += __shfl_down(s, off, 32);
    if ((tid & 31) == 0) red[tid >> 5] = s;
    __syncthreads();
    if (tid == 0) {
        float t = red[0] + red[1] + red[2] + red[3];
        t += u[0] * lin_w[HD] + w[0] * lin_w[HD + 1] + lin_b[0];
        out[0] = t;
    }
}

// ---------------- host side ----------------
extern "C" void kernel_launch(void* const* d_in, const int* in_sizes, int n_in,
                              void* d_out, int out_size, void* d_ws, size_t ws_size,
                              hipStream_t stream) {
    const float* x     = (const float*)d_in[0];
    const int*   ei    = (const int*)d_in[1];
    const float* u     = (const float*)d_in[2];
    const float* wsc   = (const float*)d_in[3];
    const float* Ws[5] = {(const float*)d_in[4],  (const float*)d_in[6],
                          (const float*)d_in[8],  (const float*)d_in[10],
                          (const float*)d_in[12]};
    const float* bs[5] = {(const float*)d_in[5],  (const float*)d_in[7],
                          (const float*)d_in[9],  (const float*)d_in[11],
                          (const float*)d_in[13]};
    const float* lin_w = (const float*)d_in[14];
    const float* lin_b = (const float*)d_in[15];

    const int n = in_sizes[0] / HD;    // 50000 nodes
    const int e = in_sizes[1] / 2;     // 1.6M edges
    const int* rowp = ei;              // sources
    const int* colp = ei + e;          // targets (aggregation index)

    // carve workspace (all within d_ws; ~90 MB total)
    char* p = (char*)d_ws;
    auto carve = [&](size_t bytes) -> void* {
        void* r = (void*)p;
        p += (bytes + 255) & ~(size_t)255;
        return r;
    };
    float*    dinv = (float*)carve((size_t)n * 4);
    float*    h    = (float*)carve((size_t)n * HD * 4);
    float*    hw   = (float*)carve((size_t)n * HD * 4);
    float*    acc  = (float*)carve((size_t)n * HD * 4);
    _Float16* h16  = (_Float16*)carve((size_t)n * HD * 2);
    _Float16* wpk  = (_Float16*)carve((size_t)HD * HD * 2);
    float*    gsum = (float*)carve((size_t)HD * 4);

    const long long nh4 = (long long)n * (HD / 4);     // float4 elements
    const int B = 256;
    dim3 blk(B);
    auto blocksLL = [](long long t, int b) { return (unsigned)((t + b - 1) / b); };

    // normalization: deg -> dinv (in place)
    k_deg_init<<<blocksLL(n, B), blk, 0, stream>>>(dinv, n);
    k_deg_count<<<blocksLL(e, B), blk, 0, stream>>>(colp, dinv, e);
    k_rsqrt_inplace<<<blocksLL(n, B), blk, 0, stream>>>(dinv, n);

    // layer 0 input: x -> f16
    k_f32_to_f16_v4<<<blocksLL(nh4, B), blk, 0, stream>>>(x, h16, nh4);

    for (int l = 0; l < 5; ++l) {
        // pack W into WMMA fragment order (f32 -> f16, 16K elements)
        k_pack_w<<<(HD * HD + B - 1) / B, blk, 0, stream>>>(Ws[l], wpk);
        // hw = h16 @ W  (WMMA, f32 accumulate)
        k_gemm_wmma<<<(n + 15) / 16, dim3(128), 0, stream>>>(h16, wpk, hw, n);
        // acc = self-loop term + bias, then scatter edge messages (f32 atomics, L2 resident)
        k_init_acc<<<blocksLL(nh4, B), blk, 0, stream>>>(hw, dinv, bs[l], acc, nh4);
        k_scatter_edges<<<blocksLL(((long long)e + 7) / 8, 1), dim3(B), 0, stream>>>(
            hw, rowp, colp, dinv, acc, e);
        // activation fused with f16 convert for the next layer
        k_leaky_dual<<<blocksLL(nh4, B), blk, 0, stream>>>(acc, h, h16, nh4);
    }

    // readout: column sums -> mean -> linear head
    k_zero128<<<1, dim3(HD), 0, stream>>>(gsum);
    k_colsum<<<(n + CHUNK - 1) / CHUNK, dim3(HD), 0, stream>>>(h, gsum, n);
    k_finalize<<<1, dim3(HD), 0, stream>>>(gsum, lin_w, lin_b, u, wsc, (float*)d_out,
                                           1.0f / (float)n);
}
